// Transformer_swap_once_38010460570047
// MI455X (gfx1250) — matrix operand order repaired and checked by reading, not compiled, not run on hardware
//
#include <hip/hip_runtime.h>
#include <hip/hip_bf16.h>

// ---------------------------------------------------------------------------
// B=2, C=512, H=W=64, L=4096, D=256, CIN=4608, k=3, pad=1.
// Compute: embed GEMMs 38.7 GF + corr GEMM 17.2 GF (bf16 WMMA, f32 accum),
// traffic ~70MB -> matrix-core bound; corr is flash-style (never materialized).
// ---------------------------------------------------------------------------
#define BB   2
#define CC   512
#define LL   4096
#define DD   256
#define CINN 4608
#define EPSF 1e-12f

typedef __attribute__((ext_vector_type(16))) __bf16 v16bf;
typedef __attribute__((ext_vector_type(8)))  float  v8f;

union FragB16 { unsigned int u[8]; uint4 q[2]; v16bf v; };
union FragF   { float f[8];        v8f   v; };
union PackBF2 { __bf16 h[2]; unsigned int u; };
union PackBF4 { __bf16 h[4]; uint2 u2; };

// ---- CDNA5 async global->LDS path (probe-confirmed builtin; typed casts) ----
#if defined(__has_builtin)
#if __has_builtin(__builtin_amdgcn_global_load_async_to_lds_b128)
#define ASYNC_LDS 1
#endif
#endif

typedef int v4i_g __attribute__((vector_size(16)));
typedef __attribute__((address_space(1))) v4i_g* gv4i_p;   // 64-bit global ptr
typedef __attribute__((address_space(3))) v4i_g* lv4i_p;   // 32-bit LDS ptr

__device__ __forceinline__ void async_copy16(void* lds_dst, const void* gsrc) {
#ifdef ASYNC_LDS
    __builtin_amdgcn_global_load_async_to_lds_b128(
        (gv4i_p)(unsigned long long)gsrc,
        (lv4i_p)(unsigned int)(unsigned long long)lds_dst,
        0, 0);
#else
    *(uint4*)lds_dst = *(const uint4*)gsrc;
#endif
}

__device__ __forceinline__ void wait_async() {
#ifdef ASYNC_LDS
#if __has_builtin(__builtin_amdgcn_s_wait_asynccnt)
    __builtin_amdgcn_s_wait_asynccnt(0);
#else
    asm volatile("s_wait_asynccnt 0" ::: "memory");
#endif
#endif
}

// ---------------------------------------------------------------------------
// Kernel 0: W_embed [D, CIN] f32 -> bf16 (native cvt, packed stores).
// ---------------------------------------------------------------------------
__global__ __launch_bounds__(256) void k_wconv(const float* __restrict__ w,
                                               unsigned short* __restrict__ wbf) {
    int i = (blockIdx.x * 256 + threadIdx.x) * 4;   // N = 1,179,648 = 1152*1024
    float4 f = *(const float4*)(w + i);
    PackBF4 o;
    o.h[0] = (__bf16)f.x; o.h[1] = (__bf16)f.y;
    o.h[2] = (__bf16)f.z; o.h[3] = (__bf16)f.w;
    *(uint2*)(wbf + i) = o.u2;
}

// ---------------------------------------------------------------------------
// Kernel 1: patch-embed GEMM with on-the-fly 3x3 unfold.
// 256 threads / 8 waves; tile 64(L) x 256(D); K = 4608 in 144 steps of 32.
// Double-buffered, bank-swizzled LDS staging; B rows DMA'd (async if present).
// The 64 L rows of a tile are one image row (l0 = 64*ltile), so each staging
// thread owns a fixed K column: one (ch, ki, kj) chain per step, 8 contiguous
// pixel loads. All fragment/staging LDS offsets are loop-invariant.
// LDS map: As[p] @ p*4096 (64x64B) | Bs[p] @ 8192+p*16384 (256x64B) |
//          Cs (f32 64x256) aliases everything after the K loop.
// ---------------------------------------------------------------------------
__global__ __launch_bounds__(256) void k_embed(const float* __restrict__ query,
                                               const float* __restrict__ key,
                                               const unsigned short* __restrict__ wbf,
                                               unsigned short* __restrict__ emb,
                                               float* __restrict__ norms) {
    const int ltile = blockIdx.x;          // image row y, and l0 = 64*ltile
    const int which = blockIdx.y;          // 0=query, 1=key
    const int b     = blockIdx.z;
    const int l0    = ltile * 64;
    const float* src = which ? key : query;

    __shared__ unsigned char smem[65536];
    float* Cs = (float*)smem;              // [64][256] after the K loop

    const int tid  = threadIdx.x;
    const int wave = tid >> 5;
    const int lane = tid & 31;
    const int hm   = lane & 15;
    const int hi   = lane >> 4;
    const int Ml   = wave & 3;
    const int ng   = (wave >> 2) * 8;

    // ---- staging state: fixed K column per thread ----
    const int kk = tid & 31;
    const int ms = (tid >> 5) * 8;
    int ch = kk / 9;
    int rr = kk - ch * 9;
    int adst[8];
#pragma unroll
    for (int e = 0; e < 8; e++) {
        int m = ms + e;   // swizzled: chunk' = (chunk + row) & 3
        adst[e] = m * 64 + ((((kk >> 3) + m) & 3) << 4) + ((kk * 2) & 15);
    }
    const unsigned short* wrow = wbf + (size_t)tid * CINN;   // thread t stages W row d=t
    int bdst[4];
#pragma unroll
    for (int j = 0; j < 4; j++) bdst[j] = tid * 64 + (((j + tid) & 3) << 4);

    // ---- loop-invariant fragment offsets (swizzled 16B chunks) ----
    const int ra = Ml * 16 + hm;
    const int aoffA0 = ra * 64 + (((hi + ra) & 3) << 4);
    const int aoffA1 = ra * 64 + (((2 + hi + ra) & 3) << 4);
    int boffB[8][2];
#pragma unroll
    for (int i = 0; i < 8; i++) {
        int rb = (ng + i) * 16 + hm;
        boffB[i][0] = rb * 64 + (((hi + rb) & 3) << 4);
        boffB[i][1] = rb * 64 + (((2 + hi + rb) & 3) << 4);
    }

    auto stageA = [&](unsigned char* Asp) {
        int kdi = (rr >= 6) ? 2 : ((rr >= 3) ? 1 : 0);
        int kdj = rr - 3 * kdi;
        int sy = ltile + kdi - 1;
        bool rowok = (unsigned)sy < 64u;
        const float* sp = src + (((size_t)b * CC + ch) << 12) + (sy << 6) + (kdj - 1);
#pragma unroll
        for (int e = 0; e < 8; e++) {
            int m = ms + e;
            bool ok = rowok && ((unsigned)(m + kdj - 1) < 64u);
            float v = ok ? sp[m] : 0.0f;
            *(__bf16*)(Asp + adst[e]) = (__bf16)v;
        }
        rr += 5;                                 // c advances by 32 = 3*9 + 5
        if (rr >= 9) { rr -= 9; ch += 4; } else { ch += 3; }
    };
    auto stageB = [&](unsigned char* Bsp, int c0) {
#pragma unroll
        for (int j = 0; j < 4; j++)
            async_copy16(Bsp + bdst[j], wrow + c0 + j * 8);
    };

    FragF acc[8];
#pragma unroll
    for (int i = 0; i < 8; i++)
#pragma unroll
        for (int r = 0; r < 8; r++) acc[i].f[r] = 0.0f;

    // prologue: stage tile 0
    stageB(smem + 8192, 0);
    stageA(smem);
    wait_async();
    __syncthreads();

    for (int it = 0; it < 144; ++it) {
        const int p = it & 1;
        const bool more = (it + 1 < 144);
        unsigned char* Asp = smem + p * 4096;
        unsigned char* Bsp = smem + 8192 + p * 16384;
        unsigned char* Asn = smem + (p ^ 1) * 4096;
        unsigned char* Bsn = smem + 8192 + (p ^ 1) * 16384;
#ifdef ASYNC_LDS
        if (more) stageB(Bsn, (it + 1) * 32);          // DMA overlaps the WMMAs
#endif
        // all fragments first, then 8 back-to-back WMMAs
        FragB16 a;
        a.q[0] = *(const uint4*)(Asp + aoffA0);
        a.q[1] = *(const uint4*)(Asp + aoffA1);
        FragB16 bb[8];
#pragma unroll
        for (int i = 0; i < 8; i++) {
            bb[i].q[0] = *(const uint4*)(Bsp + boffB[i][0]);
            bb[i].q[1] = *(const uint4*)(Bsp + boffB[i][1]);
        }
#pragma unroll
        for (int i = 0; i < 8; i++)
            acc[i].v = __builtin_amdgcn_wmma_f32_16x16x32_bf16(
                false, a.v, false, bb[i].v, (short)0, acc[i].v, false, false);
#ifndef ASYNC_LDS
        if (more) stageB(Bsn, (it + 1) * 32);
#endif
        if (more) stageA(Asn);                          // VALU co-executes with XDL
        wait_async();
        __syncthreads();
    }

    // ---- spill accumulators to Cs (f32), then norms + bf16 embedding ----
#pragma unroll
    for (int i = 0; i < 8; i++)
#pragma unroll
        for (int r = 0; r < 8; r++)
            Cs[(Ml * 16 + hi * 8 + r) * 256 + (ng + i) * 16 + hm] = acc[i].f[r];
    __syncthreads();

    if (tid < 64) {
        const float* row = Cs + tid * 256;
        float s = 0.0f;
#pragma unroll 8
        for (int d = 0; d < 256; d++) s += row[d] * row[d];
        norms[((b * 2 + which) << 12) + l0 + tid] = sqrtf(s);
    }
    unsigned int* dst = (unsigned int*)emb + ((size_t)((b * 2 + which) * LL + l0)) * 128;
#pragma unroll
    for (int j = 0; j < 32; j++) {
        int i  = j * 256 + tid;
        int m  = i >> 7;
        int dd = (i & 127) << 1;
        PackBF2 pk;
        pk.h[0] = (__bf16)Cs[m * 256 + dd];
        pk.h[1] = (__bf16)Cs[m * 256 + dd + 1];
        dst[i] = pk.u;
    }
}

// ---------------------------------------------------------------------------
// Kernel 2: flash-style correlation + max/argmax over the key axis.
// 256 threads / 8 waves. Eq tile (64x256 bf16, swizzled) LDS-resident;
// Ek streamed as double-buffered 32-row tiles (async DMA when available).
// Wave w: Ml = w>>2 (l strip of 16 within 32), Nm = w&3 (16-query subtile).
// corr = dot(Ek,Eq); scale by 1/||Ek[l]|| inside the max, 1/||Eq[m]|| at end.
// LDS map: Eqs @ 0 (64x512B) | Ek[p] @ 32768 + p*16384 (32x512B).
// ---------------------------------------------------------------------------
__global__ __launch_bounds__(256) void k_corr(const unsigned short* __restrict__ emb,
                                              const float* __restrict__ norms,
                                              float* __restrict__ S,
                                              unsigned int* __restrict__ amax) {
    const int mtile = blockIdx.x;   // 0..63
    const int b     = blockIdx.y;
    const int m0    = mtile * 64;

    __shared__ unsigned char smem[65536];
    unsigned char* Eqs = smem;              // 64 rows x 512B (swizzled)

    const int tid  = threadIdx.x;
    const int wave = tid >> 5;
    const int lane = tid & 31;
    const int hm   = lane & 15;
    const int hi   = lane >> 4;
    const int Ml   = wave >> 2;             // 0..1
    const int Nm   = wave & 3;              // 0..3

    const unsigned char* embB = (const unsigned char*)emb;
    const unsigned char* eqg = embB + ((size_t)(b * 2 * LL + m0)) * 512;
    const unsigned char* ekg = embB + ((size_t)((b * 2 + 1) * LL)) * 512;
    const float* nkk = norms + ((b * 2 + 1) << 12);
    const float* nkq = norms + ((b * 2 + 0) << 12);

    // ---- stage Eq tile (2048 chunks) + Ek tile 0 (1024 chunks), swizzled ----
#pragma unroll
    for (int n = 0; n < 8; n++) {
        int i = n * 256 + tid;
        int row = i >> 5, c = i & 31;
        async_copy16(Eqs + row * 512 + (((c + row) & 31) << 4), eqg + (size_t)i * 16);
    }
    int ssrc[4], sdst[4];
#pragma unroll
    for (int n = 0; n < 4; n++) {
        int i = n * 256 + tid;
        int row = i >> 5, c = i & 31;
        ssrc[n] = i * 16;
        sdst[n] = row * 512 + (((c + row) & 31) << 4);
        async_copy16(smem + 32768 + sdst[n], ekg + ssrc[n]);
    }
    wait_async();
    __syncthreads();

    // ---- loop-invariant fragment offsets ----
    const int ra = Ml * 16 + hm;
    const int rb = Nm * 16 + hm;
    int aoff[8][2], boff[8][2];
#pragma unroll
    for (int ks = 0; ks < 8; ks++) {
        int c1 = ks * 4 + hi, c2 = ks * 4 + 2 + hi;
        aoff[ks][0] = ra * 512 + (((c1 + ra) & 31) << 4);
        aoff[ks][1] = ra * 512 + (((c2 + ra) & 31) << 4);
        int d1 = ks * 4 + hi, d2 = d1 + 2;
        boff[ks][0] = rb * 512 + (((d1 + rb) & 31) << 4);
        boff[ks][1] = rb * 512 + (((d2 + rb) & 31) << 4);
    }

    float bestv = -1e30f;
    unsigned int besti = 0u;

    for (int l0 = 0; l0 < LL; l0 += 32) {
        const int pp = (l0 >> 5) & 1;
        unsigned char* Ekp = smem + 32768 + pp * 16384;
        unsigned char* Ekn = smem + 32768 + (pp ^ 1) * 16384;
        const bool more = (l0 + 32 < LL);
#ifdef ASYNC_LDS
        if (more) {
            const unsigned char* esrc = ekg + (size_t)(l0 + 32) * 512;
#pragma unroll
            for (int n = 0; n < 4; n++) async_copy16(Ekn + sdst[n], esrc + ssrc[n]);
        }
#endif
        FragF acc;
#pragma unroll
        for (int r = 0; r < 8; r++) acc.f[r] = 0.0f;

        // 2-deep software pipeline over the 8 K-steps
        FragB16 aC, bC, aN, bN;
        aC.q[0] = *(const uint4*)(Ekp + aoff[0][0]);
        aC.q[1] = *(const uint4*)(Ekp + aoff[0][1]);
        bC.q[0] = *(const uint4*)(Eqs + boff[0][0]);
        bC.q[1] = *(const uint4*)(Eqs + boff[0][1]);
#pragma unroll
        for (int ks = 0; ks < 8; ks++) {
            if (ks < 7) {
                aN.q[0] = *(const uint4*)(Ekp + aoff[ks + 1][0]);
                aN.q[1] = *(const uint4*)(Ekp + aoff[ks + 1][1]);
                bN.q[0] = *(const uint4*)(Eqs + boff[ks + 1][0]);
                bN.q[1] = *(const uint4*)(Eqs + boff[ks + 1][1]);
            }
            acc.v = __builtin_amdgcn_wmma_f32_16x16x32_bf16(
                false, aC.v, false, bC.v, (short)0, acc.v, false, false);
            aC = aN; bC = bN;
        }
#ifndef ASYNC_LDS
        if (more) {
            const unsigned char* esrc = ekg + (size_t)(l0 + 32) * 512;
#pragma unroll
            for (int n = 0; n < 4; n++) async_copy16(Ekn + sdst[n], esrc + ssrc[n]);
        }
#endif
        // scale rows by 1/||Ek[l]||, fold into running max (ascending l)
#pragma unroll
        for (int r = 0; r < 8; r++) {
            int row = Ml * 16 + hi * 8 + r;
            float sc = 1.0f / fmaxf(nkk[l0 + row], EPSF);
            float v = acc.f[r] * sc;
            unsigned int gl = (unsigned int)(l0 + row);
            if (v > bestv) { bestv = v; besti = gl; }
        }
        wait_async();
        __syncthreads();
    }

    // ---- final reduction (4 entries per query m), aliased onto the tiles ----
    float* redv        = (float*)smem;                 // 256 floats
    unsigned int* redi = (unsigned int*)(smem + 1024); // 256 u32
    int eidx = (Nm * 16 + hm) * 4 + Ml * 2 + hi;
    redv[eidx] = bestv;
    redi[eidx] = besti;
    __syncthreads();

    if (tid < 64) {
        float bv = -1e30f;
        unsigned int bi = 0xFFFFFFFFu;
#pragma unroll
        for (int j = 0; j < 4; j++) {
            float v = redv[tid * 4 + j];
            unsigned int ii = redi[tid * 4 + j];
            if (v > bv || (v == bv && ii < bi)) { bv = v; bi = ii; }
        }
        float nq = fmaxf(nkq[m0 + tid], EPSF);
        S[(b << 12) + m0 + tid]    = bv / nq;
        amax[(b << 12) + m0 + tid] = bi;
    }
}

// ---------------------------------------------------------------------------
// Kernel 3: swap + fold (analytic divisor cnt(y)*cnt(x); 9-tap argmax gather).
// ---------------------------------------------------------------------------
__global__ __launch_bounds__(128) void k_swapfold(const float* __restrict__ value,
                                                  const unsigned int* __restrict__ amax,
                                                  float* __restrict__ T) {
    const int l = blockIdx.x;
    const int b = blockIdx.y;
    const int y = l >> 6, x = l & 63;
    const int tid = threadIdx.x;

    int soff[9];
#pragma unroll
    for (int di = 0; di < 3; di++) {
#pragma unroll
        for (int dj = 0; dj < 3; dj++) {
            int ly = y + 1 - di, lx = x + 1 - dj;
            int o = -1;
            if ((unsigned)ly < 64u && (unsigned)lx < 64u) {
                unsigned int a = amax[(b << 12) + (ly << 6) + lx];
                int sy = (int)(a >> 6) + di - 1;
                int sx = (int)(a & 63) + dj - 1;
                if ((unsigned)sy < 64u && (unsigned)sx < 64u) o = (sy << 6) + sx;
            }
            soff[di * 3 + dj] = o;
        }
    }
    int cy = 3 - (y == 0) - (y == 63);
    int cx = 3 - (x == 0) - (x == 63);
    float inv = 1.0f / (float)(cy * cx);

    const float* vb = value + ((size_t)b << 21);
    float* tb = T + ((size_t)b << 21);
#pragma unroll
    for (int cc = 0; cc < 4; cc++) {
        int c = tid + cc * 128;
        const float* vp = vb + ((size_t)c << 12);
        float s = 0.0f;
#pragma unroll
        for (int t9 = 0; t9 < 9; t9++)
            if (soff[t9] >= 0) s += vp[soff[t9]];
        tb[((size_t)c << 12) + l] = s * inv;
    }
}

// ---------------------------------------------------------------------------
// Launch. Inputs: [0]=query f32, [1]=key f32, [2]=value f32, [3]=W_embed f32,
// [4]=k(==3). Output: S [2,1,64,64] then T [2,512,64,64], f32, concatenated.
// Workspace: wbf@0 (2,359,296B) | emb@2359296 (8,388,608B) |
//            norms@10747904 (65,536B) | amax@10813440 (32,768B).
// ---------------------------------------------------------------------------
extern "C" void kernel_launch(void* const* d_in, const int* in_sizes, int n_in,
                              void* d_out, int out_size, void* d_ws, size_t ws_size,
                              hipStream_t stream) {
    (void)in_sizes; (void)n_in; (void)out_size; (void)ws_size;
    const float* query = (const float*)d_in[0];
    const float* key   = (const float*)d_in[1];
    const float* value = (const float*)d_in[2];
    const float* wemb  = (const float*)d_in[3];

    unsigned char* ws = (unsigned char*)d_ws;
    unsigned short* wbf  = (unsigned short*)(ws);
    unsigned short* emb  = (unsigned short*)(ws + 2359296);
    float*          nrm  = (float*)(ws + 10747904);
    unsigned int*   amax = (unsigned int*)(ws + 10813440);

    float* S = (float*)d_out;
    float* T = (float*)d_out + 8192;

    k_wconv<<<1152, 256, 0, stream>>>(wemb, wbf);
    k_embed<<<dim3(64, 2, 2), 256, 0, stream>>>(query, key, wbf, emb, nrm);
    k_corr<<<dim3(64, 2), 256, 0, stream>>>(emb, nrm, S, amax);
    k_swapfold<<<dim3(4096, 2), 128, 0, stream>>>(value, amax, T);
}
